// EGNNLayer_57045755625874
// MI455X (gfx1250) — matrix operand order repaired and checked by reading, not compiled, not run on hardware
//
#include <hip/hip_runtime.h>
#include <hip/hip_bf16.h>

typedef __attribute__((ext_vector_type(4)))  _Float16 v4h;
typedef __attribute__((ext_vector_type(8)))  _Float16 v8h;
typedef __attribute__((ext_vector_type(16))) _Float16 v16h;
typedef __attribute__((ext_vector_type(8)))  float    v8f;

#define WPB 4   // waves per block (block = 128 threads, wave32)

// packed-weight layout offsets (in halves) inside d_ws
#define OFF_EW1 0        // 9 chunks * 8 tiles * 512 = 36864
#define OFF_EW2 36864    // 4 chunks -> 16384
#define OFF_NW1 53248    // 8 chunks -> 32768
#define OFF_NW2 86016    // 4 chunks -> 16384
#define PK_TOTAL 102400

__device__ __forceinline__ v4h cvt4(float4 v) {
    v4h r; r[0] = (_Float16)v.x; r[1] = (_Float16)v.y;
           r[2] = (_Float16)v.z; r[3] = (_Float16)v.w; return r;
}
__device__ __forceinline__ v16h combine(v8h lo, v8h hi) {
    return __builtin_shufflevector(lo, hi, 0,1,2,3,4,5,6,7,8,9,10,11,12,13,14,15);
}
__device__ __forceinline__ v8f wmma_f16(v16h a, v16h b, v8f c) {
    return __builtin_amdgcn_wmma_f32_16x16x32_f16(false, a, false, b, (short)0, c, false, false);
}
__device__ __forceinline__ float silu_f(float v) {
    return v / (1.0f + __expf(-v));
}

// ---------------------------------------------------------------------------
// Pack fp32 weights [K x 128] into f16 B-fragment layout:
// dst[((c*8 + t)*32 + lane)*16 + e] = W[(32c + 16*(e>>3) + 8*(lane>>4) + (e&7))*128 + 16t + (lane&15)]
// (k >= Ktot zero-padded). One wmma B operand = 16 contiguous halves per lane.
// ---------------------------------------------------------------------------
__global__ void egnn_pack_weights(const float* __restrict__ eW1,
                                  const float* __restrict__ eW2,
                                  const float* __restrict__ nW1,
                                  const float* __restrict__ nW2,
                                  _Float16* __restrict__ dst) {
    int gid = blockIdx.x * blockDim.x + threadIdx.x;
    if (gid >= PK_TOTAL) return;
    const float* W; int Ktot; int local;
    if (gid < OFF_EW2)      { W = eW1; Ktot = 272; local = gid; }
    else if (gid < OFF_NW1) { W = eW2; Ktot = 128; local = gid - OFF_EW2; }
    else if (gid < OFF_NW2) { W = nW1; Ktot = 256; local = gid - OFF_NW1; }
    else                    { W = nW2; Ktot = 128; local = gid - OFF_NW2; }
    int c    = local >> 12;
    int r    = local & 4095;
    int t    = r >> 9;
    int r2   = r & 511;
    int lane = r2 >> 4;
    int e    = r2 & 15;
    int ncol = t * 16 + (lane & 15);
    int h    = lane >> 4;
    int k    = c * 32 + (e >> 3) * 16 + h * 8 + (e & 7);
    float v  = (k < Ktot) ? W[k * 128 + ncol] : 0.0f;
    dst[gid] = (_Float16)v;
}

__global__ void egnn_zero(float4* __restrict__ p, int n4) {
    int i = blockIdx.x * blockDim.x + threadIdx.x;
    if (i < n4) p[i] = make_float4(0.f, 0.f, 0.f, 0.f);
}

// ---------------------------------------------------------------------------
// Edge MLP: per wave, 16 edges. edge_input[16 x 288(pad)] -> silu GEMM1 ->
// silu GEMM2 -> atomic scatter-add into agg[row].
// ---------------------------------------------------------------------------
__global__ __launch_bounds__(WPB * 32)
void egnn_edge_kernel(const float* __restrict__ x, const int* __restrict__ eidx,
                      const float* __restrict__ eattr,
                      const _Float16* __restrict__ w1p, const float* __restrict__ b1,
                      const _Float16* __restrict__ w2p, const float* __restrict__ b2,
                      float* __restrict__ agg, int E, int N) {
    __shared__ _Float16 s_in[WPB * 16 * 288];
    __shared__ _Float16 s_h [WPB * 16 * 128];

    const int lane = threadIdx.x & 31;
    const int wv   = threadIdx.x >> 5;
    const long tile = (long)blockIdx.x * WPB + wv;
    const long base = tile * 16;
    if (base >= E) return;                 // wave-uniform: EXEC stays all-1s

    _Float16* sm = s_in + wv * (16 * 288);
    _Float16* hm = s_h  + wv * (16 * 128);
    const int* rowI = eidx;
    const int* colI = eidx + E;

    // ---- cooperative gather: 2 lanes per edge row ----
    {
        const int m = lane >> 1, hf = lane & 1;
        long e = base + m; if (e >= E) e = E - 1;
        int ir = rowI[e]; ir = (ir < 0) ? 0 : (ir >= N ? N - 1 : ir);
        int ic = colI[e]; ic = (ic < 0) ? 0 : (ic >= N ? N - 1 : ic);
        const float4* xr = (const float4*)(x + (long)ir * 128);
        const float4* xc = (const float4*)(x + (long)ic * 128);
        _Float16* r0 = sm + m * 288 +       hf * 64;
        _Float16* r1 = sm + m * 288 + 128 + hf * 64;
#pragma unroll
        for (int q = 0; q < 16; ++q) {
            *(v4h*)(r0 + q * 4) = cvt4(xr[hf * 16 + q]);
            *(v4h*)(r1 + q * 4) = cvt4(xc[hf * 16 + q]);
        }
        const float4* ap = (const float4*)(eattr + e * 16);
#pragma unroll
        for (int q = 0; q < 2; ++q)
            *(v4h*)(sm + m * 288 + 256 + hf * 8 + q * 4) = cvt4(ap[hf * 2 + q]);
        v4h z; z[0] = z[1] = z[2] = z[3] = (_Float16)0.f;
        *(v4h*)(sm + m * 288 + 272 + hf * 8)     = z;   // pad K 272..287
        *(v4h*)(sm + m * 288 + 272 + hf * 8 + 4) = z;
    }

    // ---- A fragments for GEMM1 (16x32 f16 layout, ISA 7.12.2) ----
    const int am = lane & 15, ah = lane >> 4;
    v16h aF[9];
#pragma unroll
    for (int c = 0; c < 9; ++c) {
        v8h lo = *(const v8h*)(sm + am * 288 + c * 32 +      ah * 8);
        v8h hi = *(const v8h*)(sm + am * 288 + c * 32 + 16 + ah * 8);
        aF[c] = combine(lo, hi);
    }

    const int ncol = lane & 15, nh = lane >> 4;

    // ---- GEMM1: [16x288] @ [288x128] -> silu -> h (LDS) ----
#pragma unroll
    for (int t = 0; t < 8; ++t) {
        v8f acc = {};
#pragma unroll
        for (int c = 0; c < 9; ++c) {
            v16h bF = *(const v16h*)(w1p + ((c * 8 + t) * 32 + lane) * 16);
            acc = wmma_f16(aF[c], bF, acc);
        }
        float bias = b1[t * 16 + ncol];
#pragma unroll
        for (int j = 0; j < 8; ++j) {
            float v = silu_f(acc[j] + bias);
            hm[(j + 8 * nh) * 128 + t * 16 + ncol] = (_Float16)v;
        }
    }

    // ---- A fragments for GEMM2 from h ----
    v16h a2[4];
#pragma unroll
    for (int c = 0; c < 4; ++c) {
        v8h lo = *(const v8h*)(hm + am * 128 + c * 32 +      ah * 8);
        v8h hi = *(const v8h*)(hm + am * 128 + c * 32 + 16 + ah * 8);
        a2[c] = combine(lo, hi);
    }

    // ---- GEMM2: [16x128] @ [128x128] -> silu -> atomic scatter ----
#pragma unroll
    for (int t = 0; t < 8; ++t) {
        v8f acc = {};
#pragma unroll
        for (int c = 0; c < 4; ++c) {
            v16h bF = *(const v16h*)(w2p + ((c * 8 + t) * 32 + lane) * 16);
            acc = wmma_f16(a2[c], bF, acc);
        }
        float bias = b2[t * 16 + ncol];
#pragma unroll
        for (int j = 0; j < 8; ++j) {
            float v = silu_f(acc[j] + bias);
            int mm = j + 8 * nh;
            long e = base + mm;
            if (e < E) {
                int dr = rowI[e]; dr = (dr < 0) ? 0 : (dr >= N ? N - 1 : dr);
                unsafeAtomicAdd(agg + (long)dr * 128 + t * 16 + ncol, v);
            }
        }
    }
}

// ---------------------------------------------------------------------------
// Node MLP: per wave, 16 nodes. [x || agg] (K=256) -> silu GEMM1 -> GEMM2
// (no silu) -> out = x + update. agg aliases out (d_out); each row is fully
// consumed by its owning wave before being overwritten.
// ---------------------------------------------------------------------------
__global__ __launch_bounds__(WPB * 32)
void egnn_node_kernel(const float* __restrict__ x, const float* __restrict__ agg,
                      const _Float16* __restrict__ w1p, const float* __restrict__ b1,
                      const _Float16* __restrict__ w2p, const float* __restrict__ b2,
                      float* __restrict__ out, int N) {
    __shared__ _Float16 s_in[WPB * 16 * 256];
    __shared__ _Float16 s_h [WPB * 16 * 128];

    const int lane = threadIdx.x & 31;
    const int wv   = threadIdx.x >> 5;
    const long tile = (long)blockIdx.x * WPB + wv;
    const long base = tile * 16;
    if (base >= N) return;                 // wave-uniform

    _Float16* sm = s_in + wv * (16 * 256);
    _Float16* hm = s_h  + wv * (16 * 128);

    {
        const int m = lane >> 1, hf = lane & 1;
        long node = base + m; if (node >= N) node = N - 1;
        const float4* xr = (const float4*)(x   + node * 128);
        const float4* gr = (const float4*)(agg + node * 128);
        _Float16* r0 = sm + m * 256 +       hf * 64;
        _Float16* r1 = sm + m * 256 + 128 + hf * 64;
#pragma unroll
        for (int q = 0; q < 16; ++q) {
            *(v4h*)(r0 + q * 4) = cvt4(xr[hf * 16 + q]);
            *(v4h*)(r1 + q * 4) = cvt4(gr[hf * 16 + q]);
        }
    }

    const int am = lane & 15, ah = lane >> 4;
    v16h aF[8];
#pragma unroll
    for (int c = 0; c < 8; ++c) {
        v8h lo = *(const v8h*)(sm + am * 256 + c * 32 +      ah * 8);
        v8h hi = *(const v8h*)(sm + am * 256 + c * 32 + 16 + ah * 8);
        aF[c] = combine(lo, hi);
    }

    const int ncol = lane & 15, nh = lane >> 4;

#pragma unroll
    for (int t = 0; t < 8; ++t) {
        v8f acc = {};
#pragma unroll
        for (int c = 0; c < 8; ++c) {
            v16h bF = *(const v16h*)(w1p + ((c * 8 + t) * 32 + lane) * 16);
            acc = wmma_f16(aF[c], bF, acc);
        }
        float bias = b1[t * 16 + ncol];
#pragma unroll
        for (int j = 0; j < 8; ++j) {
            float v = silu_f(acc[j] + bias);
            hm[(j + 8 * nh) * 128 + t * 16 + ncol] = (_Float16)v;
        }
    }

    v16h a2[4];
#pragma unroll
    for (int c = 0; c < 4; ++c) {
        v8h lo = *(const v8h*)(hm + am * 128 + c * 32 +      ah * 8);
        v8h hi = *(const v8h*)(hm + am * 128 + c * 32 + 16 + ah * 8);
        a2[c] = combine(lo, hi);
    }

#pragma unroll
    for (int t = 0; t < 8; ++t) {
        v8f acc = {};
#pragma unroll
        for (int c = 0; c < 4; ++c) {
            v16h bF = *(const v16h*)(w2p + ((c * 8 + t) * 32 + lane) * 16);
            acc = wmma_f16(a2[c], bF, acc);
        }
        float bias = b2[t * 16 + ncol];
#pragma unroll
        for (int j = 0; j < 8; ++j) {
            float v = acc[j] + bias;          // NO silu on second node layer
            int mm = j + 8 * nh;
            long node = base + mm;
            if (node < N) {
                long o = node * 128 + t * 16 + ncol;
                out[o] = x[o] + v;            // fp32 residual
            }
        }
    }
}

extern "C" void kernel_launch(void* const* d_in, const int* in_sizes, int n_in,
                              void* d_out, int out_size, void* d_ws, size_t ws_size,
                              hipStream_t stream) {
    const float* x    = (const float*)d_in[0];
    const int*   eidx = (const int*)  d_in[1];
    const float* ea   = (const float*)d_in[2];
    const float* eW1  = (const float*)d_in[3];
    const float* eb1  = (const float*)d_in[4];
    const float* eW2  = (const float*)d_in[5];
    const float* eb2  = (const float*)d_in[6];
    const float* nW1  = (const float*)d_in[7];
    const float* nb1  = (const float*)d_in[8];
    const float* nW2  = (const float*)d_in[9];
    const float* nb2  = (const float*)d_in[10];
    float* out = (float*)d_out;
    _Float16* wp = (_Float16*)d_ws;

    const int N = in_sizes[0] / 128;
    const int E = in_sizes[1] / 2;

    // 1) pack weights to f16 B-fragment layout
    egnn_pack_weights<<<(PK_TOTAL + 255) / 256, 256, 0, stream>>>(eW1, eW2, nW1, nW2, wp);

    // 2) zero agg (aliased on d_out)
    int n4 = (N * 128) / 4;
    egnn_zero<<<(n4 + 255) / 256, 256, 0, stream>>>((float4*)out, n4);

    // 3) edge MLP + scatter-add into agg
    long etiles = ((long)E + 15) / 16;
    int  eblocks = (int)((etiles + WPB - 1) / WPB);
    egnn_edge_kernel<<<eblocks, WPB * 32, 0, stream>>>(
        x, eidx, ea, wp + OFF_EW1, eb1, wp + OFF_EW2, eb2, out, E, N);

    // 4) node MLP + residual (reads agg rows from d_out, then overwrites them)
    long ntiles = ((long)N + 15) / 16;
    int  nblocks = (int)((ntiles + WPB - 1) / WPB);
    egnn_node_kernel<<<nblocks, WPB * 32, 0, stream>>>(
        x, out, wp + OFF_NW1, nb1, wp + OFF_NW2, nb2, out, N);
}
